// MultiBranchDenseTDNNLayer_30159260352970
// MI455X (gfx1250) — compile-verified, matrix operand order checked by reading, not run on hardware
//
#include <hip/hip_runtime.h>
#include <math.h>

// ---- problem constants (from reference) ----
#define Bsz   16
#define CINC  1024
#define Tn    2000
#define BNC   512
#define OUTC  512
#define NBRN  3
#define KW    3
#define MAXSH 3          // max |(k-1)/2 * dil| = 1*3
#define BN_EPS  1e-5f
#define STD_EPS 0.01f

typedef unsigned short u16;
typedef unsigned int   u32;

typedef __attribute__((ext_vector_type(16))) __bf16 v16bf;
typedef __attribute__((ext_vector_type(8)))  float  v8f;

// ---- CDNA5 async global->LDS copy (ASYNCcnt), guarded so compile never breaks
#if defined(__has_builtin)
#if __has_builtin(__builtin_amdgcn_global_load_async_to_lds_b128) && \
    __has_builtin(__builtin_amdgcn_s_wait_asynccnt)
#define HAVE_ASYNC_LDS 1
#endif
#endif
#ifndef HAVE_ASYNC_LDS
#define HAVE_ASYNC_LDS 0
#endif

#if HAVE_ASYNC_LDS
// builtin prototype (from toolchain diagnostic): v4i addrspace(1)* src,
// v4i addrspace(3)* dst, imm offset, imm cpol
typedef int b128_t __attribute__((vector_size(16)));
typedef __attribute__((address_space(1))) b128_t* g_b128_p;
typedef __attribute__((address_space(3))) b128_t* l_b128_p;
#endif

__device__ __forceinline__ void cp_b128(u16* dst_lds, const u16* src_g) {
#if HAVE_ASYNC_LDS
  __builtin_amdgcn_global_load_async_to_lds_b128(
      (g_b128_p)src_g, (l_b128_p)dst_lds, 0, 0);
#else
  *(uint4*)dst_lds = *(const uint4*)src_g;
#endif
}
__device__ __forceinline__ void wait_async_barrier() {
#if HAVE_ASYNC_LDS
  __builtin_amdgcn_s_wait_asynccnt(0);
#endif
  __syncthreads();
}

__device__ __forceinline__ u16 f2bf(float f) {
  u32 u = __float_as_uint(f);
  u32 r = u + 0x7FFFu + ((u >> 16) & 1u);   // round-to-nearest-even
  return (u16)(r >> 16);
}
__device__ __forceinline__ float bf2f(u16 h) {
  return __uint_as_float(((u32)h) << 16);
}

// Load one 16x32(bf16) WMMA A/B fragment from an LDS tile stored as
// [row][32 k] (row = M for A, row = N for B-transposed). Per ISA layout:
// lanes 0-15: dwords0-3 = K0..7,  dwords4-7 = K16..23
// lanes16-31: dwords0-3 = K8..15, dwords4-7 = K24..31
__device__ __forceinline__ v16bf frag_ld(const u16* tile, int row, int khalf) {
  union U { uint4 u[2]; v16bf v; } u;
  const uint4* p = (const uint4*)(tile + row * 32 + khalf * 8);
  u.u[0] = p[0];    // K base .. +7
  u.u[1] = p[2];    // +16 elements (=+32 bytes)
  return u.v;
}

// ---------------- prep kernels ----------------
// BN1 + ReLU, store TRANSPOSED bf16: xt[b][t][c]  (so GEMM1 B-tiles are
// contiguous 64B rows, ideal for async b128 staging)
__global__ void k_bn1(const float* __restrict__ x, const float* __restrict__ g,
                      const float* __restrict__ be, const float* __restrict__ mn,
                      const float* __restrict__ vr, u16* __restrict__ xt, int n) {
  int idx = blockIdx.x * 256 + threadIdx.x;
  if (idx >= n) return;
  int t = idx % Tn;
  int c = (idx / Tn) % CINC;
  int b = idx / (Tn * CINC);
  float inv = g[c] * rsqrtf(vr[c] + BN_EPS);
  float sh  = be[c] - mn[c] * inv;
  float v = x[idx] * inv + sh;
  xt[((size_t)b * Tn + t) * CINC + c] = f2bf(v > 0.f ? v : 0.f);
}

__global__ void k_cvt(const float* __restrict__ in, u16* __restrict__ outp, int n) {
  int idx = blockIdx.x * 256 + threadIdx.x;
  if (idx < n) outp[idx] = f2bf(in[idx]);
}

// conv_w [n][o][c][k] (f32) -> wt [(n*3+k)][o][c] (bf16), K-contiguous A tiles
__global__ void k_cvtw(const float* __restrict__ cw, u16* __restrict__ wt, int n) {
  int idx = blockIdx.x * 256 + threadIdx.x;
  if (idx >= n) return;
  int k = idx % KW;
  int c = (idx / KW) % BNC;
  int o = (idx / (KW * BNC)) % OUTC;
  int nb = idx / (KW * BNC * OUTC);
  wt[(((size_t)nb * KW + k) * OUTC + o) * BNC + c] = f2bf(cw[idx]);
}

__global__ void k_bn2pre(const float* __restrict__ g, const float* __restrict__ be,
                         const float* __restrict__ mn, const float* __restrict__ vr,
                         float* __restrict__ inv2, float* __restrict__ sh2) {
  int o = blockIdx.x * 256 + threadIdx.x;
  if (o >= BNC) return;
  float inv = g[o] * rsqrtf(vr[o] + BN_EPS);
  inv2[o] = inv;
  sh2[o]  = be[o] - mn[o] * inv;
}

// ---------------- GEMM1: Ht = relu(bn2(W1 @ bn_relu(x)))^T ----------------
// C[512 x T] per batch. Block 128x128, K-step 32, 8 waves of 64x32 (4x2 WMMA).
// Double-buffered async LDS pipeline: stage tile k+1 while computing tile k.
// B rows with t >= Tn are clamped (junk feeds only unstored C columns).
__device__ __forceinline__ void stage_gemm1(u16* sA, u16* sB,
                                            const u16* Ag, const u16* Bg,
                                            int k0, int t0, int tid) {
#pragma unroll
  for (int s = 0; s < 2; s++) {
    int c = tid + s * 256;
    int row = c >> 2, kc = (c & 3) * 8;
    cp_b128(sA + row * 32 + kc, Ag + (size_t)row * CINC + k0 + kc);
  }
#pragma unroll
  for (int s = 0; s < 2; s++) {
    int c = tid + s * 256;
    int row = c >> 2, kc = (c & 3) * 8;
    int t = t0 + row;
    t = t < Tn ? t : Tn - 1;                 // branch-free edge handling
    cp_b128(sB + row * 32 + kc, Bg + (size_t)t * CINC + k0 + kc);
  }
}

__global__ __launch_bounds__(256) void k_gemm1(const u16* __restrict__ W1b,
                                               const u16* __restrict__ xt,
                                               const float* __restrict__ inv2,
                                               const float* __restrict__ sh2,
                                               u16* __restrict__ Ht) {
  __shared__ u16 sA[2][128 * 32];
  __shared__ u16 sB[2][128 * 32];   // B transposed: [n(t)][k(c)]
  const int tid = threadIdx.x;
  const int lane = tid & 31, wave = tid >> 5;
  const int wm = wave >> 2, wn = wave & 3;
  const int r = lane & 15, kh = lane >> 4;
  const int b  = blockIdx.z;
  const int m0 = blockIdx.y * 128;
  const int t0 = blockIdx.x * 128;

  v8f acc[4][2];
#pragma unroll
  for (int mt = 0; mt < 4; mt++)
#pragma unroll
    for (int nt = 0; nt < 2; nt++)
#pragma unroll
      for (int e = 0; e < 8; e++) acc[mt][nt][e] = 0.f;

  const u16* Ag = W1b + (size_t)m0 * CINC;
  const u16* Bg = xt + (size_t)b * Tn * CINC;

  stage_gemm1(sA[0], sB[0], Ag, Bg, 0, t0, tid);
  int cur = 0;
  for (int k0 = 0; k0 < CINC; k0 += 32) {
    wait_async_barrier();                     // buf[cur] resident; all waves past buf[cur^1]
    if (k0 + 32 < CINC)
      stage_gemm1(sA[cur ^ 1], sB[cur ^ 1], Ag, Bg, k0 + 32, t0, tid);

    v16bf afr[4], bfr[2];
#pragma unroll
    for (int mt = 0; mt < 4; mt++) afr[mt] = frag_ld(sA[cur] + (wm * 64 + mt * 16) * 32, r, kh);
#pragma unroll
    for (int nt = 0; nt < 2; nt++) bfr[nt] = frag_ld(sB[cur] + (wn * 32 + nt * 16) * 32, r, kh);
#pragma unroll
    for (int mt = 0; mt < 4; mt++)
#pragma unroll
      for (int nt = 0; nt < 2; nt++)
        acc[mt][nt] = __builtin_amdgcn_wmma_f32_16x16x32_bf16(
            false, afr[mt], false, bfr[nt], (short)0, acc[mt][nt], false, false);
    cur ^= 1;
  }

  // epilogue: BN2 + ReLU -> bf16 Ht[b][t][o]
#pragma unroll
  for (int mt = 0; mt < 4; mt++)
#pragma unroll
    for (int nt = 0; nt < 2; nt++) {
      int tcol = t0 + wn * 32 + nt * 16 + r;
#pragma unroll
      for (int i = 0; i < 8; i++) {
        int o = m0 + wm * 64 + mt * 16 + i + 8 * kh;
        float v = acc[mt][nt][i] * inv2[o] + sh2[o];
        v = v > 0.f ? v : 0.f;
        if (tcol < Tn) Ht[((size_t)b * Tn + tcol) * BNC + o] = f2bf(v);
      }
    }
}

// ---------------- conv branches: f[branch] = dilated conv as 3 shifted GEMMs ----
__device__ __forceinline__ void stage_conv(u16* sA, u16* sB,
                                           const u16* Aw, const u16* Hb,
                                           int k0, int t0, int tid) {
#pragma unroll
  for (int s = 0; s < 6; s++) {
    int c = tid + s * 256;
    int j = c >> 9;
    int rr = c & 511;
    int row = rr >> 2, kc = (rr & 3) * 8;
    cp_b128(sA + j * 128 * 32 + row * 32 + kc,
            Aw + ((size_t)j * OUTC + row) * BNC + k0 + kc);
  }
  // halo columns must be true zeros (conv 'same' padding)
  for (int idx = tid; idx < (128 + 2 * MAXSH) * 4; idx += 256) {
    int n = idx >> 2, kc = (idx & 3) * 8;
    int tg = t0 + n - MAXSH;
    if (tg >= 0 && tg < Tn)
      cp_b128(sB + n * 32 + kc, Hb + (size_t)tg * BNC + k0 + kc);
    else {
      uint4 z = {0u, 0u, 0u, 0u};
      *(uint4*)(sB + n * 32 + kc) = z;
    }
  }
}

__global__ __launch_bounds__(256) void k_conv(const u16* __restrict__ wt,
                                              const u16* __restrict__ Ht,
                                              u16* __restrict__ F) {
  __shared__ u16 sA[2][3 * 128 * 32];
  __shared__ u16 sB[2][(128 + 2 * MAXSH) * 32];
  const int tid = threadIdx.x;
  const int lane = tid & 31, wave = tid >> 5;
  const int wm = wave >> 2, wn = wave & 3;
  const int r = lane & 15, kh = lane >> 4;
  const int branch = blockIdx.z >> 4;   // 0..2  (dil = branch+1)
  const int b  = blockIdx.z & 15;
  const int m0 = blockIdx.y * 128;
  const int t0 = blockIdx.x * 128;
  const int dil = branch + 1;

  v8f acc[4][2];
#pragma unroll
  for (int mt = 0; mt < 4; mt++)
#pragma unroll
    for (int nt = 0; nt < 2; nt++)
#pragma unroll
      for (int e = 0; e < 8; e++) acc[mt][nt][e] = 0.f;

  const u16* Hb = Ht + (size_t)b * Tn * BNC;
  const u16* Aw = wt + ((size_t)branch * KW * OUTC + m0) * BNC;

  stage_conv(sA[0], sB[0], Aw, Hb, 0, t0, tid);
  int cur = 0;
  for (int k0 = 0; k0 < BNC; k0 += 32) {
    wait_async_barrier();
    if (k0 + 32 < BNC)
      stage_conv(sA[cur ^ 1], sB[cur ^ 1], Aw, Hb, k0 + 32, t0, tid);

#pragma unroll
    for (int j = 0; j < 3; j++) {
      int shift = MAXSH + (j - 1) * dil;   // 0..6, always in-halo
      v16bf afr[4], bfr[2];
#pragma unroll
      for (int mt = 0; mt < 4; mt++)
        afr[mt] = frag_ld(sA[cur] + j * 128 * 32 + (wm * 64 + mt * 16) * 32, r, kh);
#pragma unroll
      for (int nt = 0; nt < 2; nt++)
        bfr[nt] = frag_ld(sB[cur] + (wn * 32 + nt * 16 + shift) * 32, r, kh);
#pragma unroll
      for (int mt = 0; mt < 4; mt++)
#pragma unroll
        for (int nt = 0; nt < 2; nt++)
          acc[mt][nt] = __builtin_amdgcn_wmma_f32_16x16x32_bf16(
              false, afr[mt], false, bfr[nt], (short)0, acc[mt][nt], false, false);
    }
    cur ^= 1;
  }

#pragma unroll
  for (int mt = 0; mt < 4; mt++)
#pragma unroll
    for (int nt = 0; nt < 2; nt++) {
      int tcol = t0 + wn * 32 + nt * 16 + r;
#pragma unroll
      for (int i = 0; i < 8; i++) {
        int o = m0 + wm * 64 + mt * 16 + i + 8 * kh;
        if (tcol < Tn)
          F[(((size_t)branch * Bsz + b) * OUTC + o) * Tn + tcol] = f2bf(acc[mt][nt][i]);
      }
    }
}

// ---------------- stats pooling: mean/std/skew/kurt over T of sum-of-branches ----
__global__ __launch_bounds__(256) void k_stats(const u16* __restrict__ F,
                                               float* __restrict__ stats) {
  __shared__ float r0[256], r1[256], r2[256], r3[256];
  const int tid = threadIdx.x;
  const int row = blockIdx.x;
  const int b = row >> 9, o = row & 511;
  const size_t base = ((size_t)b * OUTC + o) * Tn;
  const size_t stride = (size_t)Bsz * OUTC * Tn;
  float s1 = 0.f, s2 = 0.f, s3 = 0.f, s4 = 0.f;
  for (int t = tid; t < Tn; t += 256) {
    float v = bf2f(F[base + t]) + bf2f(F[base + stride + t]) + bf2f(F[base + 2 * stride + t]);
    float v2 = v * v;
    s1 += v; s2 += v2; s3 += v2 * v; s4 += v2 * v2;
  }
  r0[tid] = s1; r1[tid] = s2; r2[tid] = s3; r3[tid] = s4;
  __syncthreads();
  for (int st = 128; st > 0; st >>= 1) {
    if (tid < st) {
      r0[tid] += r0[tid + st]; r1[tid] += r1[tid + st];
      r2[tid] += r2[tid + st]; r3[tid] += r3[tid + st];
    }
    __syncthreads();
  }
  if (tid == 0) {
    float n = (float)Tn;
    float S1 = r0[0], S2 = r1[0], S3 = r2[0], S4 = r3[0];
    float mu = S1 / n;
    float var = (S2 - n * mu * mu) / (n - 1.f);
    var = var > 0.f ? var : 0.f;
    float sd = sqrtf(var);
    float sc = sd > STD_EPS ? sd : STD_EPS;
    float m3 = S3 / n - 3.f * mu * (S2 / n) + 2.f * mu * mu * mu;
    float m4 = S4 / n - 4.f * mu * (S3 / n) + 6.f * mu * mu * (S2 / n) - 3.f * mu * mu * mu * mu;
    float* st = stats + (size_t)b * 4 * OUTC;
    st[o] = mu;
    st[OUTC + o] = sd;
    st[2 * OUTC + o] = m3 / (sc * sc * sc);
    st[3 * OUTC + o] = m4 / (sc * sc * sc * sc);
  }
}

// ---------------- selection MLP ----------------
__global__ void k_z(const float* __restrict__ stats, const float* __restrict__ sw1,
                    const float* __restrict__ sb1, float* __restrict__ z) {
  int idx = blockIdx.x * 256 + threadIdx.x;
  if (idx >= Bsz * OUTC) return;
  int b = idx >> 9, r = idx & 511;
  const float* st = stats + (size_t)b * 4 * OUTC;
  const float* w = sw1 + (size_t)r * 4 * OUTC;
  float a = 0.f;
  for (int i = 0; i < 4 * OUTC; i++) a += st[i] * w[i];
  z[idx] = a + sb1[r];
}

__global__ void k_sel(const float* __restrict__ z, const float* __restrict__ sw2,
                      const float* __restrict__ sb2, float* __restrict__ sw) {
  int idx = blockIdx.x * 256 + threadIdx.x;
  if (idx >= Bsz * OUTC) return;
  int b = idx >> 9, o = idx & 511;
  const float* zz = z + (size_t)b * OUTC;
  float l[NBRN];
#pragma unroll
  for (int n = 0; n < NBRN; n++) {
    const float* w = sw2 + ((size_t)n * OUTC + o) * OUTC;
    float a = 0.f;
    for (int r = 0; r < OUTC; r++) a += zz[r] * w[r];
    l[n] = a + sb2[n * OUTC + o];
  }
  float mx = fmaxf(l[0], fmaxf(l[1], l[2]));
  float e0 = expf(l[0] - mx), e1 = expf(l[1] - mx), e2 = expf(l[2] - mx);
  float s = e0 + e1 + e2;
  sw[((size_t)b * NBRN + 0) * OUTC + o] = e0 / s;
  sw[((size_t)b * NBRN + 1) * OUTC + o] = e1 / s;
  sw[((size_t)b * NBRN + 2) * OUTC + o] = e2 / s;
}

// ---------------- final weighted branch sum ----------------
__global__ void k_final(const u16* __restrict__ F, const float* __restrict__ sw,
                        float* __restrict__ out, int n) {
  int idx = blockIdx.x * 256 + threadIdx.x;
  if (idx >= n) return;
  int b = idx / (OUTC * Tn);
  int o = (idx / Tn) % OUTC;
  size_t stride = (size_t)Bsz * OUTC * Tn;
  float a = 0.f;
#pragma unroll
  for (int nb = 0; nb < NBRN; nb++)
    a += bf2f(F[(size_t)nb * stride + idx]) * sw[((size_t)b * NBRN + nb) * OUTC + o];
  out[idx] = a;
}

// ---------------- launch ----------------
extern "C" void kernel_launch(void* const* d_in, const int* in_sizes, int n_in,
                              void* d_out, int out_size, void* d_ws, size_t ws_size,
                              hipStream_t stream) {
  (void)in_sizes; (void)n_in; (void)out_size; (void)ws_size;
  const float* x   = (const float*)d_in[0];
  const float* g1  = (const float*)d_in[1];
  const float* be1 = (const float*)d_in[2];
  const float* mn1 = (const float*)d_in[3];
  const float* vr1 = (const float*)d_in[4];
  const float* W1  = (const float*)d_in[5];
  const float* g2  = (const float*)d_in[6];
  const float* be2 = (const float*)d_in[7];
  const float* mn2 = (const float*)d_in[8];
  const float* vr2 = (const float*)d_in[9];
  const float* cw  = (const float*)d_in[10];
  const float* sw1 = (const float*)d_in[11];
  const float* sb1 = (const float*)d_in[12];
  const float* sw2 = (const float*)d_in[13];
  const float* sb2 = (const float*)d_in[14];
  float* out = (float*)d_out;

  unsigned char* ws = (unsigned char*)d_ws;
  size_t off = 0;
  auto carve = [&](size_t bytes) -> void* {
    void* p = ws + off;
    off += (bytes + 255) & ~(size_t)255;
    return p;
  };
  u16*   xt    = (u16*)  carve((size_t)Bsz * CINC * Tn * 2);   // [b][t][c] bf16
  u16*   w1b   = (u16*)  carve((size_t)BNC * CINC * 2);
  u16*   Ht    = (u16*)  carve((size_t)Bsz * Tn * BNC * 2);    // [b][t][o] bf16
  u16*   wt    = (u16*)  carve((size_t)NBRN * KW * OUTC * BNC * 2);
  u16*   Fh    = (u16*)  carve((size_t)NBRN * Bsz * OUTC * Tn * 2);
  float* stats = (float*)carve((size_t)Bsz * 4 * OUTC * 4);
  float* zbuf  = (float*)carve((size_t)Bsz * OUTC * 4);
  float* swt   = (float*)carve((size_t)Bsz * NBRN * OUTC * 4);
  float* inv2  = (float*)carve((size_t)BNC * 4);
  float* sh2   = (float*)carve((size_t)BNC * 4);

  int n_x = Bsz * CINC * Tn;
  k_bn1<<<(n_x + 255) / 256, 256, 0, stream>>>(x, g1, be1, mn1, vr1, xt, n_x);
  int n_w1 = BNC * CINC;
  k_cvt<<<(n_w1 + 255) / 256, 256, 0, stream>>>(W1, w1b, n_w1);
  int n_cw = NBRN * OUTC * BNC * KW;
  k_cvtw<<<(n_cw + 255) / 256, 256, 0, stream>>>(cw, wt, n_cw);
  k_bn2pre<<<2, 256, 0, stream>>>(g2, be2, mn2, vr2, inv2, sh2);

  dim3 gg1((Tn + 127) / 128, BNC / 128, Bsz);
  k_gemm1<<<gg1, 256, 0, stream>>>(w1b, xt, inv2, sh2, Ht);

  dim3 ggc((Tn + 127) / 128, OUTC / 128, NBRN * Bsz);
  k_conv<<<ggc, 256, 0, stream>>>(wt, Ht, Fh);

  k_stats<<<Bsz * OUTC, 256, 0, stream>>>(Fh, stats);
  k_z<<<(Bsz * OUTC + 255) / 256, 256, 0, stream>>>(stats, sw1, sb1, zbuf);
  k_sel<<<(Bsz * OUTC + 255) / 256, 256, 0, stream>>>(zbuf, sw2, sb2, swt);

  int n_out = Bsz * OUTC * Tn;
  k_final<<<(n_out + 255) / 256, 256, 0, stream>>>(Fh, swt, out, n_out);
}
